// DeepSeekMoE_76845554860062
// MI455X (gfx1250) — matrix operand order repaired and checked by reading, not compiled, not run on hardware
//
#include <hip/hip_runtime.h>
#include <hip/hip_bf16.h>

#define HH   2048   // hidden
#define MM   1408   // intermediate
#define EE   8      // routed experts
#define NEXP 9      // routed + shared
#define NTOK 1024   // B*S

typedef __attribute__((ext_vector_type(16))) __bf16 v16bf;
typedef __attribute__((ext_vector_type(8)))  float  v8f;

__device__ __forceinline__ unsigned short f2bf(float f) {
  unsigned u = __float_as_uint(f);
  u += 0x7fffu + ((u >> 16) & 1u);          // round-to-nearest-even
  return (unsigned short)(u >> 16);
}

__device__ __forceinline__ v8f wmma_bf16(v16bf a, v16bf b, v8f c) {
  return __builtin_amdgcn_wmma_f32_16x16x32_bf16(
      false, a, false, b, (short)0, c, false, false);
}

// ---- A fragment (16 rows x 32 K) from bf16 row-major [rows, ld] ----
// lanes 0-15: rows 0-15, K {0..7,16..23}; lanes 16-31: rows 0-15, K {8..15,24..31}
__device__ __forceinline__ v16bf load_fragA_bf16(const unsigned short* base, int ld,
                                                 int row0, int k) {
  int lane = threadIdx.x & 31;
  const unsigned short* p =
      base + (size_t)(row0 + (lane & 15)) * ld + k + ((lane >> 4) << 3);
  union { uint4 q[2]; v16bf b; } fr;
  fr.q[0] = *(const uint4*)(p);        // K kb0..kb0+7
  fr.q[1] = *(const uint4*)(p + 16);   // K kb0+16..kb0+23
  return fr.b;
}

// ---- B fragment (32 K x 16 cols); B[k][c] = W[(col0+c)*ld + k], W bf16 row-major ----
// lane picks column (lane&15); lanes 0-15 hold K {0..7,16..23}, lanes 16-31 hold K {8..15,24..31}
__device__ __forceinline__ v16bf load_fragB_bf16(const unsigned short* base, int ld,
                                                 int col0, int k) {
  int lane = threadIdx.x & 31;
  const unsigned short* p =
      base + (size_t)(col0 + (lane & 15)) * ld + k + ((lane >> 4) << 3);
  __builtin_prefetch(p + 64, 0, 1);    // 2 K-steps ahead on the streaming weight row
  union { uint4 q[2]; v16bf b; } fr;
  fr.q[0] = *(const uint4*)(p);
  fr.q[1] = *(const uint4*)(p + 16);
  return fr.b;
}

// ============  kernel A: bulk fp32 -> bf16 conversion (bandwidth-bound)  ============
__global__ void __launch_bounds__(256)
cvt_f32_bf16(const float* __restrict__ src, unsigned short* __restrict__ dst, long n) {
  long i = ((long)blockIdx.x * 256 + threadIdx.x) * 8;
  if (i >= n) return;
  float4 f0 = *(const float4*)(src + i);
  float4 f1 = *(const float4*)(src + i + 4);
  uint4 o;
  o.x = (unsigned)f2bf(f0.x) | ((unsigned)f2bf(f0.y) << 16);
  o.y = (unsigned)f2bf(f0.z) | ((unsigned)f2bf(f0.w) << 16);
  o.z = (unsigned)f2bf(f1.x) | ((unsigned)f2bf(f1.y) << 16);
  o.w = (unsigned)f2bf(f1.z) | ((unsigned)f2bf(f1.w) << 16);
  *(uint4*)(dst + i) = o;
}

// =======================  kernel 0: zero aux stats  =======================
__global__ void moe_init_stats(float* stats) {
  if (threadIdx.x < 16) stats[threadIdx.x] = 0.0f;
}

// =======================  kernel 1: router/gate  ==========================
__global__ void __launch_bounds__(256)
moe_gate(const float* __restrict__ x, const float* __restrict__ gw,
         float* __restrict__ cw, float* __restrict__ stats) {
  int n = blockIdx.x * 256 + threadIdx.x;
  if (n >= NTOK) return;
  float lg[EE];
#pragma unroll
  for (int e = 0; e < EE; ++e) lg[e] = 0.0f;
  const float* xr = x + (size_t)n * HH;
  for (int h = 0; h < HH; ++h) {
    float xv = xr[h];
#pragma unroll
    for (int e = 0; e < EE; ++e) lg[e] += xv * gw[e * HH + h];
  }
  float mx = lg[0];
#pragma unroll
  for (int e = 1; e < EE; ++e) mx = fmaxf(mx, lg[e]);
  float sc[EE], sum = 0.0f;
#pragma unroll
  for (int e = 0; e < EE; ++e) { sc[e] = __expf(lg[e] - mx); sum += sc[e]; }
  float inv = 1.0f / sum;
#pragma unroll
  for (int e = 0; e < EE; ++e) sc[e] *= inv;
  int i1 = 0;
#pragma unroll
  for (int e = 1; e < EE; ++e) if (sc[e] > sc[i1]) i1 = e;
  int i2 = (i1 == 0) ? 1 : 0;
#pragma unroll
  for (int e = 0; e < EE; ++e) if (e != i1 && sc[e] > sc[i2]) i2 = e;
  float w1 = sc[i1], w2 = sc[i2];
  float nrm = 1.0f / (w1 + w2 + 1e-20f);
#pragma unroll
  for (int e = 0; e < EE; ++e) cw[n * NEXP + e] = 0.0f;
  cw[n * NEXP + i1] = w1 * nrm;
  cw[n * NEXP + i2] = w2 * nrm;
  cw[n * NEXP + EE] = 1.0f;                 // shared expert
#pragma unroll
  for (int e = 0; e < EE; ++e) atomicAdd(&stats[e], sc[e]);
  atomicAdd(&stats[EE + i1], 1.0f);
  atomicAdd(&stats[EE + i2], 1.0f);
}

// ============  kernel 2: gate/up GEMMs + SwiGLU + cw scale (all bf16 feeds) ============
// grid (MM/128, NTOK/64, NEXP), block 128 (4 waves); wave tile = 64(N) x 32(M)
__global__ void __launch_bounds__(128)
moe_swiglu(const unsigned short* __restrict__ xb,
           const unsigned short* __restrict__ wgb,   // [NEXP, MM, HH] bf16
           const unsigned short* __restrict__ wub,   // [NEXP, MM, HH] bf16
           const float* __restrict__ cw, unsigned short* __restrict__ cwa) {
  int wave = threadIdx.x >> 5;
  int m0 = (blockIdx.x * 4 + wave) * 32;
  int n0 = blockIdx.y * 64;
  int e  = blockIdx.z;
  const unsigned short* wg = wgb + (size_t)e * MM * HH;
  const unsigned short* wu = wub + (size_t)e * MM * HH;

  v8f z = {0,0,0,0,0,0,0,0};
  v8f g00=z,g01=z,g10=z,g11=z,g20=z,g21=z,g30=z,g31=z;
  v8f u00=z,u01=z,u10=z,u11=z,u20=z,u21=z,u30=z,u31=z;

  for (int k = 0; k < HH; k += 32) {
    v16bf a0 = load_fragA_bf16(xb, HH, n0,      k);
    v16bf a1 = load_fragA_bf16(xb, HH, n0 + 16, k);
    v16bf a2 = load_fragA_bf16(xb, HH, n0 + 32, k);
    v16bf a3 = load_fragA_bf16(xb, HH, n0 + 48, k);
    v16bf bg0 = load_fragB_bf16(wg, HH, m0,      k);
    v16bf bg1 = load_fragB_bf16(wg, HH, m0 + 16, k);
    v16bf bu0 = load_fragB_bf16(wu, HH, m0,      k);
    v16bf bu1 = load_fragB_bf16(wu, HH, m0 + 16, k);
    g00 = wmma_bf16(a0, bg0, g00);  g01 = wmma_bf16(a0, bg1, g01);
    g10 = wmma_bf16(a1, bg0, g10);  g11 = wmma_bf16(a1, bg1, g11);
    g20 = wmma_bf16(a2, bg0, g20);  g21 = wmma_bf16(a2, bg1, g21);
    g30 = wmma_bf16(a3, bg0, g30);  g31 = wmma_bf16(a3, bg1, g31);
    u00 = wmma_bf16(a0, bu0, u00);  u01 = wmma_bf16(a0, bu1, u01);
    u10 = wmma_bf16(a1, bu0, u10);  u11 = wmma_bf16(a1, bu1, u11);
    u20 = wmma_bf16(a2, bu0, u20);  u21 = wmma_bf16(a2, bu1, u21);
    u30 = wmma_bf16(a3, bu0, u30);  u31 = wmma_bf16(a3, bu1, u31);
  }

  int lane = threadIdx.x & 31;
  int rhi  = (lane >> 4) << 3;
  int colb = lane & 15;
  unsigned short* outa = cwa + (size_t)e * NTOK * MM;
#define EPI(GA, UA, RO, CO)                                              \
  {                                                                      \
    _Pragma("unroll") for (int r = 0; r < 8; ++r) {                      \
      int row = n0 + (RO) + r + rhi;                                     \
      int col = m0 + (CO) + colb;                                        \
      float gv = (GA)[r], uv = (UA)[r];                                  \
      float s  = gv / (1.0f + __expf(-gv));                              \
      float wt = cw[row * NEXP + e];                                     \
      outa[(size_t)row * MM + col] = f2bf(s * uv * wt);                  \
    }                                                                    \
  }
  EPI(g00, u00, 0, 0)   EPI(g01, u01, 0, 16)
  EPI(g10, u10, 16, 0)  EPI(g11, u11, 16, 16)
  EPI(g20, u20, 32, 0)  EPI(g21, u21, 32, 16)
  EPI(g30, u30, 48, 0)  EPI(g31, u31, 48, 16)
#undef EPI
}

// ==========  kernel 3: down-projection, accumulated over 9 experts  ==========
// grid (HH/128, NTOK/64), block 128 (4 waves); wave tile = 64(N) x 32(H)
__global__ void __launch_bounds__(128)
moe_down(const unsigned short* __restrict__ cwa,
         const unsigned short* __restrict__ wdb,    // [NEXP, HH, MM] bf16
         float* __restrict__ y) {
  int wave = threadIdx.x >> 5;
  int h0 = (blockIdx.x * 4 + wave) * 32;
  int n0 = blockIdx.y * 64;

  v8f z = {0,0,0,0,0,0,0,0};
  v8f c00=z,c01=z,c10=z,c11=z,c20=z,c21=z,c30=z,c31=z;

  for (int e = 0; e < NEXP; ++e) {
    const unsigned short* a  = cwa + (size_t)e * NTOK * MM;
    const unsigned short* wd = wdb + (size_t)e * HH * MM;
    for (int k = 0; k < MM; k += 32) {
      v16bf a0 = load_fragA_bf16(a, MM, n0,      k);
      v16bf a1 = load_fragA_bf16(a, MM, n0 + 16, k);
      v16bf a2 = load_fragA_bf16(a, MM, n0 + 32, k);
      v16bf a3 = load_fragA_bf16(a, MM, n0 + 48, k);
      v16bf b0 = load_fragB_bf16(wd, MM, h0,      k);
      v16bf b1 = load_fragB_bf16(wd, MM, h0 + 16, k);
      c00 = wmma_bf16(a0, b0, c00);  c01 = wmma_bf16(a0, b1, c01);
      c10 = wmma_bf16(a1, b0, c10);  c11 = wmma_bf16(a1, b1, c11);
      c20 = wmma_bf16(a2, b0, c20);  c21 = wmma_bf16(a2, b1, c21);
      c30 = wmma_bf16(a3, b0, c30);  c31 = wmma_bf16(a3, b1, c31);
    }
  }

  int lane = threadIdx.x & 31;
  int rhi  = (lane >> 4) << 3;
  int colb = lane & 15;
#define OUT(CA, RO, CO)                                                  \
  {                                                                      \
    _Pragma("unroll") for (int r = 0; r < 8; ++r)                        \
      y[(size_t)(n0 + (RO) + r + rhi) * HH + h0 + (CO) + colb] = (CA)[r];\
  }
  OUT(c00, 0, 0)   OUT(c01, 0, 16)
  OUT(c10, 16, 0)  OUT(c11, 16, 16)
  OUT(c20, 32, 0)  OUT(c21, 32, 16)
  OUT(c30, 48, 0)  OUT(c31, 48, 16)
#undef OUT
}

// =======================  kernel 4: aux loss scalar  =======================
__global__ void moe_aux(const float* __restrict__ stats, float* __restrict__ out) {
  if (threadIdx.x == 0) {
    float a = 0.0f;
#pragma unroll
    for (int e = 0; e < EE; ++e) {
      float ce = stats[EE + e] / 256.0f;    // counts / (S*K/E)
      float pi = stats[e] / (float)NTOK;    // mean score
      a += ce * pi;
    }
    *out = a * 0.001f;                      // AUX_ALPHA
  }
}

extern "C" void kernel_launch(void* const* d_in, const int* in_sizes, int n_in,
                              void* d_out, int out_size, void* d_ws, size_t ws_size,
                              hipStream_t stream) {
  const float* x       = (const float*)d_in[0];
  const float* gate_w  = (const float*)d_in[1];
  const float* w_gate  = (const float*)d_in[2];
  const float* w_up    = (const float*)d_in[3];
  const float* w_down  = (const float*)d_in[4];
  const float* sh_gate = (const float*)d_in[5];
  const float* sh_up   = (const float*)d_in[6];
  const float* sh_down = (const float*)d_in[7];
  float* y = (float*)d_out;

  // ---- workspace layout (bf16 copies of everything the GEMMs touch) ----
  char* w = (char*)d_ws;
  float*          stats = (float*)w;                               // 16 f32
  float*          cw    = (float*)(w + 256);                       // NTOK*NEXP f32
  unsigned short* xb    = (unsigned short*)(w + 65536);            // NTOK*HH
  unsigned short* cwa   = xb  + (size_t)NTOK * HH;                 // NEXP*NTOK*MM
  unsigned short* wgb   = cwa + (size_t)NEXP * NTOK * MM;          // NEXP*MM*HH
  unsigned short* wub   = wgb + (size_t)NEXP * MM * HH;            // NEXP*MM*HH
  unsigned short* wdb   = wub + (size_t)NEXP * MM * HH;            // NEXP*HH*MM

  // ---- one-shot fp32 -> bf16 conversions (streaming, HBM-bound) ----
  struct CvtJob { const float* s; unsigned short* d; long n; };
  const CvtJob jobs[7] = {
      {x,       xb,                               (long)NTOK * HH},
      {w_gate,  wgb,                              (long)EE * MM * HH},
      {sh_gate, wgb + (size_t)EE * MM * HH,       (long)MM * HH},
      {w_up,    wub,                              (long)EE * MM * HH},
      {sh_up,   wub + (size_t)EE * MM * HH,       (long)MM * HH},
      {w_down,  wdb,                              (long)EE * HH * MM},
      {sh_down, wdb + (size_t)EE * HH * MM,       (long)MM * HH},
  };
  for (int j = 0; j < 7; ++j) {
    long blocks = (jobs[j].n / 8 + 255) / 256;
    cvt_f32_bf16<<<dim3((unsigned)blocks), dim3(256), 0, stream>>>(
        jobs[j].s, jobs[j].d, jobs[j].n);
  }

  moe_init_stats<<<dim3(1), dim3(32), 0, stream>>>(stats);
  moe_gate<<<dim3(NTOK / 256), dim3(256), 0, stream>>>(x, gate_w, cw, stats);
  moe_swiglu<<<dim3(MM / 128, NTOK / 64, NEXP), dim3(128), 0, stream>>>(
      xb, wgb, wub, cw, cwa);
  moe_down<<<dim3(HH / 128, NTOK / 64), dim3(128), 0, stream>>>(cwa, wdb, y);
  moe_aux<<<dim3(1), dim3(32), 0, stream>>>(stats, y + (size_t)NTOK * HH);
}
